// GATNet_2731599200906
// MI455X (gfx1250) — compile-verified, matrix operand order checked by reading, not run on hardware
//
#include <hip/hip_runtime.h>
#include <math.h>

typedef __attribute__((ext_vector_type(16))) _Float16 v16h;
typedef __attribute__((ext_vector_type(8)))  float    v8f;
typedef int v4i __attribute__((vector_size(16)));   // matches builtin's V4i param type

#define N_NODES 20000
#define N_EDGES 320000
#define N_TOT_E 340000
#define N_GRAPH 8

#define AS_GLOBAL __attribute__((address_space(1)))
#define AS_LOCAL  __attribute__((address_space(3)))

#if defined(__has_builtin)
#if __has_builtin(__builtin_amdgcn_global_load_async_to_lds_b128)
#define HAVE_ASYNC_LDS 1
#endif
#endif
#ifndef HAVE_ASYNC_LDS
#define HAVE_ASYNC_LDS 0
#endif

__device__ __forceinline__ void wait_asynccnt0() {
#if defined(__has_builtin)
#if __has_builtin(__builtin_amdgcn_s_wait_asynccnt)
    __builtin_amdgcn_s_wait_asynccnt(0);
    return;
#else
    asm volatile("s_wait_asynccnt 0" ::: "memory");
    return;
#endif
#else
    asm volatile("s_wait_asynccnt 0" ::: "memory");
#endif
}

// ---------- small device helpers ----------
__device__ __forceinline__ float eluf(float x) { return x > 0.f ? x : (expf(x) - 1.f); }
__device__ __forceinline__ float lrelu(float x) { return x > 0.f ? x : 0.2f * x; }
// monotonic float<->uint mapping so segment-max can use global_atomic_max_u32
__device__ __forceinline__ unsigned f2ord(float f) {
    unsigned u = __float_as_uint(f);
    return (u >> 31) ? ~u : (u | 0x80000000u);
}
__device__ __forceinline__ float ord2f(unsigned s) {
    unsigned u = (s >> 31) ? (s & 0x7FFFFFFFu) : ~s;
    return __uint_as_float(u);
}

// ---------- fills ----------
__global__ void fill_f32(float* p, float v, int n) {
    int i = blockIdx.x * blockDim.x + threadIdx.x;
    if (i < n) p[i] = v;
}
__global__ void fill_u32(unsigned* p, unsigned v, int n) {
    int i = blockIdx.x * blockDim.x + threadIdx.x;
    if (i < n) p[i] = v;
}

// ---------- weight pre-swizzle into WMMA B-fragment layout (f16) ----------
// Unit u = ((cb*KT + kt)*4 + t)*32 + lane holds the v16h B fragment for
// column block cb (64 cols), k-tile kt (32 K), col subtile t (16 cols), lane.
__global__ void pack_wb(const float* __restrict__ W, v16h* __restrict__ WB,
                        int IN, int OUT) {
    const int KT = (IN + 31) / 32;
    const int units = (OUT / 64) * KT * 4 * 32;
    int u = blockIdx.x * blockDim.x + threadIdx.x;
    if (u >= units) return;
    const int lane = u & 31;
    const int t    = (u >> 5) & 3;
    const int kt   = (u >> 7) % KT;
    const int cb   = (u >> 7) / KT;
    const int m    = lane & 15;
    const int hk   = (lane >> 4) * 8;
    const int col  = cb * 64 + t * 16 + m;
    v16h hv;
#pragma unroll
    for (int i = 0; i < 8; ++i) {
        int ka = kt * 32 + hk + i;
        int kb = ka + 16;
        hv[i]     = (ka < IN) ? (_Float16)W[(size_t)ka * OUT + col] : (_Float16)0.0f;
        hv[8 + i] = (kb < IN) ? (_Float16)W[(size_t)kb * OUT + col] : (_Float16)0.0f;
    }
    WB[u] = hv;
}

// ---------- WMMA GEMM: H[N,out_total] = X[N,IN] @ W[IN,out_total] ----------
// Packed f16 weights staged into LDS via the CDNA5 async-to-LDS path
// (ASYNCcnt; no VGPR round-trip), then consumed as aligned v16h fragments.
template <int IN>
__global__ __launch_bounds__(128) void gemm_wmma(const float* __restrict__ X,
                                                 const v16h* __restrict__ WB,
                                                 float* __restrict__ H,
                                                 int n_nodes, int out_total) {
    constexpr int KT = (IN + 31) / 32;
    __shared__ v16h wb[KT * 4 * 32];

    // cooperative stage of this column-block's packed weights into LDS
    {
        const uint4* src = (const uint4*)(WB + (size_t)blockIdx.y * (KT * 4 * 32));
        uint4* dst = (uint4*)wb;
#if HAVE_ASYNC_LDS
#pragma unroll
        for (int i = threadIdx.x; i < KT * 4 * 32 * 2; i += 128) {
            __builtin_amdgcn_global_load_async_to_lds_b128(
                (AS_GLOBAL v4i*)(src + i), (AS_LOCAL v4i*)(dst + i), 0, 0);
        }
        wait_asynccnt0();
#else
#pragma unroll
        for (int i = threadIdx.x; i < KT * 4 * 32 * 2; i += 128) dst[i] = src[i];
#endif
    }
    __syncthreads();

    const int lane = threadIdx.x & 31;
    const int wave = threadIdx.x >> 5;
    const int tile = blockIdx.x * 4 + wave;   // wave-uniform
    const int row0 = tile * 16;
    if (row0 >= n_nodes) return;              // uniform exit: EXEC all-ones for WMMA
    const int col0 = blockIdx.y * 64;

    const int m  = lane & 15;
    const int hk = (lane >> 4) * 8;
    const float4* xr4 = (const float4*)(X + (size_t)(row0 + m) * IN);

    v8f acc0 = {}, acc1 = {}, acc2 = {}, acc3 = {};

#pragma unroll
    for (int kt = 0; kt < KT; ++kt) {
        const int k0 = kt * 32;
        float4 p0 = xr4[(k0 + hk) >> 2];
        float4 p1 = xr4[((k0 + hk) >> 2) + 1];
        float4 p2 = {0.f, 0.f, 0.f, 0.f}, p3 = {0.f, 0.f, 0.f, 0.f};
        if (IN >= k0 + 32) {                 // constant-folded per unrolled kt
            p2 = xr4[(k0 + hk + 16) >> 2];
            p3 = xr4[((k0 + hk + 16) >> 2) + 1];
        }
        v16h a;
        a[0]  = (_Float16)p0.x; a[1]  = (_Float16)p0.y;
        a[2]  = (_Float16)p0.z; a[3]  = (_Float16)p0.w;
        a[4]  = (_Float16)p1.x; a[5]  = (_Float16)p1.y;
        a[6]  = (_Float16)p1.z; a[7]  = (_Float16)p1.w;
        a[8]  = (_Float16)p2.x; a[9]  = (_Float16)p2.y;
        a[10] = (_Float16)p2.z; a[11] = (_Float16)p2.w;
        a[12] = (_Float16)p3.x; a[13] = (_Float16)p3.y;
        a[14] = (_Float16)p3.z; a[15] = (_Float16)p3.w;

#pragma unroll
        for (int t = 0; t < 4; ++t) {
            v16h b = wb[(kt * 4 + t) * 32 + lane];
            v8f& acc = (t == 0) ? acc0 : (t == 1) ? acc1 : (t == 2) ? acc2 : acc3;
            acc = __builtin_amdgcn_wmma_f32_16x16x32_f16(false, a, false, b, (short)0, acc,
                                                         false, false);
        }
    }

    // C layout: VGPR r -> row (r + 8*(lane>>4)), col (lane&15)
    const int srow = row0 + (lane >> 4) * 8;
    const int scol = col0 + (lane & 15);
#pragma unroll
    for (int r = 0; r < 8; ++r) {
        float* hp = H + (size_t)(srow + r) * out_total;
        hp[scol + 0]  = acc0[r];
        hp[scol + 16] = acc1[r];
        hp[scol + 32] = acc2[r];
        hp[scol + 48] = acc3[r];
    }
}

// ---------- per-node attention logits: AS/AD[n,h] = <H[n,h,:], a_src/dst[h,:]> ----------
__global__ void att_logits(const float* __restrict__ H, const float* __restrict__ a_src,
                           const float* __restrict__ a_dst, float* __restrict__ AS,
                           float* __restrict__ AD, int n_nodes, int heads, int ch) {
    int idx = blockIdx.x * blockDim.x + threadIdx.x;
    if (idx >= n_nodes * heads) return;
    int h = idx % heads, n = idx / heads;
    const float4* hp = (const float4*)(H + (size_t)n * heads * ch + h * ch);
    const float4* av = (const float4*)(a_src + h * ch);
    const float4* bv = (const float4*)(a_dst + h * ch);
    float s = 0.f, d = 0.f;
    for (int c = 0; c < ch / 4; ++c) {
        float4 v = hp[c], a4 = av[c], b4 = bv[c];
        s += v.x * a4.x + v.y * a4.y + v.z * a4.z + v.w * a4.w;
        d += v.x * b4.x + v.y * b4.y + v.z * b4.z + v.w * b4.w;
    }
    AS[idx] = s;
    AD[idx] = d;
}

// ---------- edge pass 1: segment max of leaky-relu logits over dst ----------
__global__ void edge_max(const int* __restrict__ srcI, const int* __restrict__ dstI,
                         const float* __restrict__ AS, const float* __restrict__ AD,
                         unsigned* __restrict__ EMAX, int n_edges, int n_nodes, int heads) {
    int idx = blockIdx.x * blockDim.x + threadIdx.x;
    int total = (n_edges + n_nodes) * heads;
    if (idx >= total) return;
    int h = idx % heads, e = idx / heads;
    int s = (e < n_edges) ? srcI[e] : (e - n_edges);
    int d = (e < n_edges) ? dstI[e] : (e - n_edges);
    float el = lrelu(AS[s * heads + h] + AD[d * heads + h]);
    atomicMax(&EMAX[d * heads + h], f2ord(el));
}

// ---------- edge pass 2: ee = exp(el - max); scatter denom and numerator ----------
// (softmax normalization deferred to finalize: out = num/den)
__global__ void edge_scatter(const int* __restrict__ srcI, const int* __restrict__ dstI,
                             const float* __restrict__ AS, const float* __restrict__ AD,
                             const unsigned* __restrict__ EMAX, float* __restrict__ DEN,
                             float* __restrict__ NUM, const float* __restrict__ H,
                             int n_edges, int n_nodes, int heads, int ch) {
    int idx = blockIdx.x * blockDim.x + threadIdx.x;
    int total = (n_edges + n_nodes) * heads;
    if (idx >= total) return;
    int h = idx % heads, e = idx / heads;
    int s = (e < n_edges) ? srcI[e] : (e - n_edges);
    int d = (e < n_edges) ? dstI[e] : (e - n_edges);
    const float4* hs = (const float4*)(H + (size_t)s * heads * ch + h * ch);
    __builtin_prefetch(hs, 0, 1);            // global_prefetch_b8: hide gather latency
    float el = lrelu(AS[s * heads + h] + AD[d * heads + h]);
    float ee = __expf(el - ord2f(EMAX[d * heads + h]));
    atomicAdd(&DEN[d * heads + h], ee);
    float* np = NUM + (size_t)d * heads * ch + h * ch;
    for (int c = 0; c < ch / 4; ++c) {
        float4 v = hs[c];
        atomicAdd(&np[c * 4 + 0], ee * v.x);
        atomicAdd(&np[c * 4 + 1], ee * v.y);
        atomicAdd(&np[c * 4 + 2], ee * v.z);
        atomicAdd(&np[c * 4 + 3], ee * v.w);
    }
}

// ---------- finalize: concat heads (layers 1-3) ----------
__global__ void finalize_concat(const float* __restrict__ NUM, const float* __restrict__ DEN,
                                const float* __restrict__ bias, float* __restrict__ OUT,
                                int n_nodes, int heads, int ch) {
    int idx = blockIdx.x * blockDim.x + threadIdx.x;
    int ot = heads * ch;
    if (idx >= n_nodes * ot) return;
    int n = idx / ot, rem = idx % ot, h = rem / ch;
    float v = NUM[idx] / DEN[n * heads + h] + bias[rem];
    OUT[idx] = eluf(v);
}

// ---------- finalize: mean over heads (layer 4: 4 heads x 64 ch) ----------
__global__ void finalize_mean(const float* __restrict__ NUM, const float* __restrict__ DEN,
                              const float* __restrict__ bias, float* __restrict__ OUT,
                              int n_nodes) {
    int idx = blockIdx.x * blockDim.x + threadIdx.x;
    if (idx >= n_nodes * 64) return;
    int n = idx / 64, c = idx % 64;
    float acc = 0.f;
#pragma unroll
    for (int h = 0; h < 4; ++h) acc += NUM[(size_t)n * 256 + h * 64 + c] / DEN[n * 4 + h];
    OUT[idx] = eluf(acc * 0.25f + bias[c]);
}

// ---------- per-graph pooling over function nodes ----------
__global__ void pool_scatter(const int* __restrict__ fidx, const int* __restrict__ flag,
                             const float* __restrict__ H, float* __restrict__ FP) {
    int idx = blockIdx.x * blockDim.x + threadIdx.x;
    if (idx >= 8000 * 64) return;
    int i = idx / 64, c = idx % 64;
    atomicAdd(&FP[flag[i] * 64 + c], H[(size_t)fidx[i] * 64 + c]);
}

// ---------- head MLP: fp@Wp, tgt@Wt, elu(concat) @ Wo + bo ----------
__global__ __launch_bounds__(256) void head_mlp(const float* __restrict__ FP,
                                                const float* __restrict__ H,
                                                const int* __restrict__ dvi,
                                                const float* __restrict__ Wp,
                                                const float* __restrict__ Wt,
                                                const float* __restrict__ Wo,
                                                const float* __restrict__ bo,
                                                float* __restrict__ out) {
    __shared__ float z[N_GRAPH * 128];
    int t = threadIdx.x;
    for (int w = t; w < N_GRAPH * 64; w += 256) {
        int b = w >> 6, c = w & 63;
        const float* hb = H + (size_t)dvi[b] * 64;
        float ap = 0.f, at = 0.f;
        for (int k = 0; k < 64; ++k) {
            ap += FP[b * 64 + k] * Wp[k * 64 + c];
            at += hb[k] * Wt[k * 64 + c];
        }
        z[b * 128 + c]      = eluf(ap);
        z[b * 128 + 64 + c] = eluf(at);
    }
    __syncthreads();
    if (t < N_GRAPH) {
        float acc = bo[0];
        for (int j = 0; j < 128; ++j) acc += z[t * 128 + j] * Wo[j];
        out[t] = acc;
    }
}

extern "C" void kernel_launch(void* const* d_in, const int* in_sizes, int n_in,
                              void* d_out, int out_size, void* d_ws, size_t ws_size,
                              hipStream_t stream) {
    (void)in_sizes; (void)n_in; (void)out_size; (void)ws_size;
    const int N = N_NODES, E = N_EDGES, ET = N_TOT_E;

    const float* x    = (const float*)d_in[0];
    const int*   ei   = (const int*)d_in[1];
    const int*   fidx = (const int*)d_in[2];
    const int*   flag = (const int*)d_in[3];
    const int*   dvi  = (const int*)d_in[4];
    const float* Wl[4]  = {(const float*)d_in[5],  (const float*)d_in[9],
                           (const float*)d_in[13], (const float*)d_in[17]};
    const float* Asl[4] = {(const float*)d_in[6],  (const float*)d_in[10],
                           (const float*)d_in[14], (const float*)d_in[18]};
    const float* Adl[4] = {(const float*)d_in[7],  (const float*)d_in[11],
                           (const float*)d_in[15], (const float*)d_in[19]};
    const float* Bl[4]  = {(const float*)d_in[8],  (const float*)d_in[12],
                           (const float*)d_in[16], (const float*)d_in[20]};
    const float* Wp = (const float*)d_in[21];
    const float* Wt = (const float*)d_in[22];
    const float* Wo = (const float*)d_in[23];
    const float* bo = (const float*)d_in[24];

    const int* srcI = ei;
    const int* dstI = ei + E;

    // workspace layout (floats)
    float*    XH   = (float*)d_ws;            // [N,64] layer output / next input
    float*    HW   = XH + (size_t)N * 64;     // [N,256] post-GEMM features (max)
    float*    NUM  = HW + (size_t)N * 256;    // [N,256] scatter numerator (max)
    float*    AS   = NUM + (size_t)N * 256;   // [N,8]
    float*    AD   = AS + (size_t)N * 8;      // [N,8]
    unsigned* EMAX = (unsigned*)(AD + (size_t)N * 8);  // [N,8]
    float*    DEN  = (float*)(EMAX + (size_t)N * 8);   // [N,8]
    float*    FP   = DEN + (size_t)N * 8;              // [8,64]
    v16h*     WBU  = (v16h*)(FP + 512);                // packed weights, <=32KB

    auto cd = [](int a, int b) { return (a + b - 1) / b; };

    const float* cur = x;
    int inch = 16;
    for (int l = 0; l < 4; ++l) {
        const int heads = (l == 3) ? 4 : 8;
        const int ch    = (l == 3) ? 64 : 8;
        const int ot    = heads * ch;
        const int KT    = (inch + 31) / 32;
        const int units = (ot / 64) * KT * 4 * 32;

        pack_wb<<<cd(units, 128), 128, 0, stream>>>(Wl[l], WBU, inch, ot);

        dim3 gg(cd(N / 16, 4), ot / 64);
        if (inch == 16)
            gemm_wmma<16><<<gg, 128, 0, stream>>>(cur, WBU, HW, N, ot);
        else
            gemm_wmma<64><<<gg, 128, 0, stream>>>(cur, WBU, HW, N, ot);

        att_logits<<<cd(N * heads, 256), 256, 0, stream>>>(HW, Asl[l], Adl[l], AS, AD, N,
                                                           heads, ch);
        fill_u32<<<cd(N * heads, 256), 256, 0, stream>>>(EMAX, 0x007FFFFFu /* ord(-inf) */,
                                                         N * heads);
        fill_f32<<<cd(N * heads, 256), 256, 0, stream>>>(DEN, 0.f, N * heads);
        fill_f32<<<cd(N * ot, 256), 256, 0, stream>>>(NUM, 0.f, N * ot);

        edge_max<<<cd(ET * heads, 256), 256, 0, stream>>>(srcI, dstI, AS, AD, EMAX, E, N,
                                                          heads);
        edge_scatter<<<cd(ET * heads, 256), 256, 0, stream>>>(srcI, dstI, AS, AD, EMAX, DEN,
                                                              NUM, HW, E, N, heads, ch);
        if (l < 3)
            finalize_concat<<<cd(N * ot, 256), 256, 0, stream>>>(NUM, DEN, Bl[l], XH, N,
                                                                 heads, ch);
        else
            finalize_mean<<<cd(N * 64, 256), 256, 0, stream>>>(NUM, DEN, Bl[l], XH, N);

        cur = XH;
        inch = 64;
    }

    fill_f32<<<cd(N_GRAPH * 64, 256), 256, 0, stream>>>(FP, 0.f, N_GRAPH * 64);
    pool_scatter<<<cd(8000 * 64, 256), 256, 0, stream>>>(fidx, flag, XH, FP);
    head_mlp<<<1, 256, 0, stream>>>(FP, XH, dvi, Wp, Wt, Wo, bo, (float*)d_out);
}
